// NonLocalBlock_38989713113551
// MI455X (gfx1250) — matrix operand order repaired and checked by reading, not compiled, not run on hardware
//
#include <hip/hip_runtime.h>
#include <hip/hip_bf16.h>
#include <stdint.h>

#define B_   8
#define N_   3136          // 56*56
#define CIN  256
#define CI   128
#define COUT 256
#define QT   64            // query rows per block
#define KT   64            // key tile

typedef __attribute__((ext_vector_type(16))) __bf16       v16bf;
typedef __attribute__((ext_vector_type(8)))  float        v8f;
typedef __attribute__((ext_vector_type(4)))  unsigned int u32x4;
typedef __attribute__((ext_vector_type(4)))  int          i32x4;
typedef __attribute__((ext_vector_type(8)))  int          i32x8;

union Frag {
    v16bf  v;
    __bf16 h[16];
    u32x4  q[2];
};

__device__ __forceinline__ v8f vzero8() {
    v8f z = {0.f,0.f,0.f,0.f,0.f,0.f,0.f,0.f};
    return z;
}

__device__ __forceinline__ v8f wmma_bf16(const Frag& a, const Frag& b, v8f c) {
    return __builtin_amdgcn_wmma_f32_16x16x32_bf16(
        false, a.v, false, b.v, (short)0, c, false, false);
}

// ---------------------------------------------------------------------------
// Tensor Data Mover: 2D tile (tile0 contiguous elems x tile1 rows) of bf16
// from global to LDS.  D# per CDNA5 ISA ch.10.7 / 8.3-8.4.
// ---------------------------------------------------------------------------
#if __has_builtin(__builtin_amdgcn_tensor_load_to_lds)
#define HAVE_TDM 1
__device__ __forceinline__ void tdm_load_2d_bf16(
    uint32_t lds_off, const void* gaddr,
    uint32_t tensor_d0, uint32_t tensor_d1,
    uint32_t tile_d0, uint32_t tile_d1, uint64_t stride0)
{
    uint64_t ga = (uint64_t)(uintptr_t)gaddr;
    u32x4 g0;
    g0[0] = 1u;                                   // count=1, user descriptor
    g0[1] = lds_off;                              // lds_addr (bytes)
    g0[2] = (uint32_t)ga;                         // global_addr[31:0]
    g0[3] = (uint32_t)((ga >> 32) & 0x01FFFFFFu)  // global_addr[56:32]
          | (2u << 30);                           // type = 2 ("image")
    i32x8 g1;
    g1[0] = (int)(1u << 16);                      // data_size = 1 -> 2 bytes
    g1[1] = (int)((tensor_d0 & 0xFFFFu) << 16);   // tensor_dim0[15:0] @ bits63:48
    g1[2] = (int)((tensor_d0 >> 16) | ((tensor_d1 & 0xFFFFu) << 16));
    g1[3] = (int)((tensor_d1 >> 16) | ((tile_d0 & 0xFFFFu) << 16));
    g1[4] = (int)(tile_d1 & 0xFFFFu);             // tile_dim1; tile_dim2 = 0
    g1[5] = (int)(uint32_t)stride0;               // tensor_dim0_stride[31:0]
    g1[6] = (int)((uint32_t)(stride0 >> 32) & 0xFFFFu);
    g1[7] = 0;                                    // dim1_stride unused (2D tile)
    i32x4 z4 = {0,0,0,0};
#if defined(__clang_major__) && (__clang_major__ >= 23)
    i32x8 z8 = {0,0,0,0,0,0,0,0};
    __builtin_amdgcn_tensor_load_to_lds(g0, g1, z4, z4, z8, 0);
#else
    __builtin_amdgcn_tensor_load_to_lds(g0, g1, z4, z4, 0);
#endif
}
#else
#define HAVE_TDM 0
#endif

// ---------------------------------------------------------------------------
// Kernel 1: projections.  theta/phi = bf16(X@W+b) row-major [B,N,CI];
// g is written TRANSPOSED: gT[b][chan][key]  so k_attn's F@g B-fragments are
// contiguous per lane (no LDS transpose in the hot loop).
// grid (49, B, 3), block 128 (4 waves, each wave: 16 rows x 128 cols)
// ---------------------------------------------------------------------------
__global__ __launch_bounds__(128) void k_proj(
    const float* __restrict__ X,
    const float* __restrict__ W0, const float* __restrict__ W1, const float* __restrict__ W2,
    const float* __restrict__ Bi0, const float* __restrict__ Bi1, const float* __restrict__ Bi2,
    __bf16* __restrict__ O0, __bf16* __restrict__ O1, __bf16* __restrict__ O2)
{
    __shared__ __align__(16) __bf16 sWt[CI][40];   // W chunk transposed [col][k]

    const int z = blockIdx.z;
    const float* W  = (z == 0) ? W0 : (z == 1) ? W1 : W2;
    const float* Bi = (z == 0) ? Bi0 : (z == 1) ? Bi1 : Bi2;
    __bf16* O       = (z == 0) ? O0 : (z == 1) ? O1 : O2;

    const int bi = blockIdx.y;
    const int q0 = blockIdx.x * QT;
    const int tid = threadIdx.x;
    const int wave = tid >> 5, lane = tid & 31;
    const int lr = lane & 15, hi = lane >> 4;

    v8f acc[8];
    #pragma unroll
    for (int j = 0; j < 8; ++j) acc[j] = vzero8();

    const float* Xrow = X + ((size_t)bi * N_ + q0 + wave * 16 + lr) * CIN;

    for (int kc = 0; kc < CIN / 32; ++kc) {
        const int k0 = kc * 32;
        __builtin_prefetch(Xrow + k0 + 64, 0, 0);   // chunk after next
        __syncthreads();
        {   // stage W[k0..k0+31][0..127] transposed into LDS (coalesced reads)
            const int c = tid;
            #pragma unroll 4
            for (int k = 0; k < 32; ++k)
                sWt[c][k] = (__bf16)W[(size_t)(k0 + k) * CI + c];
        }
        __syncthreads();

        // A-fragment 16x32 from X (fp32 -> bf16), documented 16-bit A layout
        Frag a;
        {
            const int kh = hi * 8;
            const float* p0 = Xrow + k0 + kh;
            const float* p1 = Xrow + k0 + 16 + kh;
            #pragma unroll
            for (int i = 0; i < 8; ++i) a.h[i]     = (__bf16)p0[i];
            #pragma unroll
            for (int i = 0; i < 8; ++i) a.h[8 + i] = (__bf16)p1[i];
        }
        #pragma unroll
        for (int j = 0; j < 8; ++j) {
            Frag b;
            const __bf16* wp = &sWt[j * 16 + lr][hi * 8];
            b.q[0] = *(const u32x4*)(wp);
            b.q[1] = *(const u32x4*)(wp + 16);
            acc[j] = wmma_bf16(a, b, acc[j]);
        }
    }

    #pragma unroll
    for (int j = 0; j < 8; ++j) {
        const int col = j * 16 + lr;
        const float bv = Bi[col];
        #pragma unroll
        for (int r = 0; r < 8; ++r) {
            const int row = q0 + wave * 16 + r + hi * 8;
            const __bf16 v = (__bf16)(acc[j][r] + bv);
            if (z < 2) O[(size_t)(bi * N_ + row) * CI + col] = v;       // row-major
            else       O[((size_t)bi * CI + col) * N_ + row] = v;       // gT scatter
        }
    }
}

// ---------------------------------------------------------------------------
// Kernel 2: flash-attention  y = softmax(theta @ phi^T) @ g
// grid (49, B), block 128 (4 waves x 16 query rows), 49 key tiles of 64.
// Double-buffered TDM pipeline: phi tile [64key][128ch] and gT tile
// [128ch][64key] are DMA'd to LDS; all GEMM fragments come from LDS.
// Dynamic LDS: 2*16KB (phi) + 2*16KB (gT) + 8KB (P staging) = 72KB (<320KB/WGP)
// ---------------------------------------------------------------------------
#define TILE_E (KT * CI)          // 8192 elements per tile buffer

__global__ __launch_bounds__(128) void k_attn(
    const __bf16* __restrict__ TH, const __bf16* __restrict__ PH,
    const __bf16* __restrict__ GT, __bf16* __restrict__ Y)
{
    extern __shared__ __align__(16) __bf16 smem[];
    __bf16* phiB = smem;                    // [2][TILE_E]  row-major [key][chan]
    __bf16* gB   = smem + 2 * TILE_E;       // [2][TILE_E]  row-major [chan][key]
    __bf16* sPa  = smem + 4 * TILE_E;       // [4][16*KT]   per-wave P staging

    const int bi = blockIdx.y;
    const int q0 = blockIdx.x * QT;
    const int tid = threadIdx.x;
    const int wave = tid >> 5, lane = tid & 31;
    const int lr = lane & 15, hi = lane >> 4;

    const __bf16* Pb  = PH + (size_t)bi * N_ * CI;    // phi[b]  [N][CI]
    const __bf16* GTb = GT + (size_t)bi * CI * N_;    // gT[b]   [CI][N]

    // Preload theta A-fragments (loop invariant): 4 chunks of K=32
    const __bf16* th = TH + ((size_t)bi * N_ + q0 + wave * 16 + lr) * CI;
    Frag at[4];
    #pragma unroll
    for (int c = 0; c < 4; ++c) {
        const int kh = c * 32 + hi * 8;
        at[c].q[0] = *(const u32x4*)(th + kh);
        at[c].q[1] = *(const u32x4*)(th + kh + 16);
    }

    v8f acc[8];
    #pragma unroll
    for (int j = 0; j < 8; ++j) acc[j] = vzero8();
    float m_[8], l_[8];
    #pragma unroll
    for (int r = 0; r < 8; ++r) { m_[r] = -3.0e38f; l_[r] = 0.f; }

    __bf16* sPw = sPa + wave * (16 * KT);
    const int NT = N_ / KT;   // 49

#if HAVE_TDM
    if (wave == 0) {   // prologue: DMA tile 0 into buffer 0
        tdm_load_2d_bf16((uint32_t)(uintptr_t)phiB, Pb,
                         CI, N_, CI, KT, CI);
        tdm_load_2d_bf16((uint32_t)(uintptr_t)gB, GTb,
                         N_, CI, KT, CI, N_);
    }
#endif

    for (int kt = 0; kt < NT; ++kt) {
        const int key0 = kt * KT;
        const int cur = kt & 1;
        const __bf16* ph_t = phiB + cur * TILE_E;
        const __bf16* g_t  = gB   + cur * TILE_E;

#if HAVE_TDM
        if (wave == 0) __builtin_amdgcn_s_wait_tensorcnt(0); // tile kt landed
        __syncthreads();    // tile visible to all; prev-iter buffer reads done
        if (wave == 0 && kt + 1 < NT) {     // prefetch tile kt+1 (other buffer)
            const int nk = (kt + 1) * KT;
            const int nb = (kt + 1) & 1;
            tdm_load_2d_bf16((uint32_t)(uintptr_t)(phiB + nb * TILE_E),
                             Pb + (size_t)nk * CI, CI, N_, CI, KT, CI);
            tdm_load_2d_bf16((uint32_t)(uintptr_t)(gB + nb * TILE_E),
                             GTb + nk, N_, CI, KT, CI, N_);
        }
#else
        __syncthreads();
        for (int idx = tid; idx < TILE_E; idx += 128)
            phiB[cur * TILE_E + idx] = Pb[(size_t)key0 * CI + idx];
        for (int idx = tid; idx < TILE_E; idx += 128)
            gB[cur * TILE_E + idx] =
                GTb[(size_t)(idx >> 6) * N_ + key0 + (idx & 63)];
        __syncthreads();
#endif

        // S = theta @ phi^T : B-frags from LDS phi tile (row-major [key][chan])
        v8f St[4];
        #pragma unroll
        for (int jj = 0; jj < 4; ++jj) {
            v8f s = vzero8();
            #pragma unroll
            for (int c = 0; c < 4; ++c) {
                Frag bp;
                const __bf16* pp = ph_t + (jj * 16 + lr) * CI + c * 32 + hi * 8;
                bp.q[0] = *(const u32x4*)(pp);
                bp.q[1] = *(const u32x4*)(pp + 16);
                s = wmma_bf16(at[c], bp, s);
            }
            St[jj] = s;
        }

        // Online softmax per row (rows per-VGPR; cols across 16 lanes, wave32)
        #pragma unroll
        for (int r = 0; r < 8; ++r) {
            float mx = fmaxf(fmaxf(St[0][r], St[1][r]), fmaxf(St[2][r], St[3][r]));
            mx = fmaxf(mx, __shfl_xor(mx, 1, 32));
            mx = fmaxf(mx, __shfl_xor(mx, 2, 32));
            mx = fmaxf(mx, __shfl_xor(mx, 4, 32));
            mx = fmaxf(mx, __shfl_xor(mx, 8, 32));
            const float mn = fmaxf(m_[r], mx);
            const float sc = __expf(m_[r] - mn);
            float ps = 0.f;
            #pragma unroll
            for (int jj = 0; jj < 4; ++jj) {
                const float p = __expf(St[jj][r] - mn);
                St[jj][r] = p;
                ps += p;
            }
            ps += __shfl_xor(ps, 1, 32);
            ps += __shfl_xor(ps, 2, 32);
            ps += __shfl_xor(ps, 4, 32);
            ps += __shfl_xor(ps, 8, 32);
            l_[r] = l_[r] * sc + ps;
            m_[r] = mn;
            #pragma unroll
            for (int j = 0; j < 8; ++j) acc[j][r] = acc[j][r] * sc;
        }

        // P: D-layout -> A-layout via per-wave LDS region
        #pragma unroll
        for (int jj = 0; jj < 4; ++jj)
            #pragma unroll
            for (int r = 0; r < 8; ++r)
                sPw[(r + hi * 8) * KT + jj * 16 + lr] = (__bf16)St[jj][r];
        asm volatile("s_wait_dscnt 0" ::: "memory");

        Frag pa[2];
        #pragma unroll
        for (int kc = 0; kc < 2; ++kc) {
            const __bf16* pp = sPw + lr * KT + kc * 32 + hi * 8;
            pa[kc].q[0] = *(const u32x4*)(pp);
            pa[kc].q[1] = *(const u32x4*)(pp + 16);
        }
        // Y += P @ G : B-frags from LDS gT tile (row-major [chan][key])
        #pragma unroll
        for (int kc = 0; kc < 2; ++kc)
            #pragma unroll
            for (int j = 0; j < 8; ++j) {
                Frag bg;
                const __bf16* gp = g_t + (j * 16 + lr) * KT + kc * 32 + hi * 8;
                bg.q[0] = *(const u32x4*)(gp);
                bg.q[1] = *(const u32x4*)(gp + 16);
                acc[j] = wmma_bf16(pa[kc], bg, acc[j]);
            }
    }

    // normalize and write y (bf16, row-major)
    #pragma unroll
    for (int j = 0; j < 8; ++j) {
        const int col = j * 16 + lr;
        #pragma unroll
        for (int r = 0; r < 8; ++r) {
            const int row = q0 + wave * 16 + r + hi * 8;
            Y[(size_t)(bi * N_ + row) * CI + col] = (__bf16)(acc[j][r] / l_[r]);
        }
    }
}

// ---------------------------------------------------------------------------
// Kernel 3: out = x + y @ W_out + b_out   (fp32 output)
// grid (49, B), block 256 (8 waves: 4 row groups x 2 column halves)
// ---------------------------------------------------------------------------
__global__ __launch_bounds__(256) void k_out(
    const __bf16* __restrict__ Yb, const float* __restrict__ Wo,
    const float* __restrict__ Bo, const float* __restrict__ X,
    float* __restrict__ OUT)
{
    __shared__ __align__(16) __bf16 sWt[COUT][40];

    const int bi = blockIdx.y;
    const int q0 = blockIdx.x * QT;
    const int tid = threadIdx.x;
    const int wave = tid >> 5, lane = tid & 31;
    const int lr = lane & 15, hi = lane >> 4;
    const int rg = wave & 3, ch = wave >> 2;

    v8f acc[8];
    #pragma unroll
    for (int j = 0; j < 8; ++j) acc[j] = vzero8();

    const __bf16* yrow = Yb + ((size_t)bi * N_ + q0 + rg * 16 + lr) * CI;

    for (int kc = 0; kc < CI / 32; ++kc) {
        const int k0 = kc * 32;
        __syncthreads();
        {
            const int c = tid;   // 0..255
            #pragma unroll 4
            for (int k = 0; k < 32; ++k)
                sWt[c][k] = (__bf16)Wo[(size_t)(k0 + k) * COUT + c];
        }
        __syncthreads();

        Frag a;
        a.q[0] = *(const u32x4*)(yrow + k0 + hi * 8);
        a.q[1] = *(const u32x4*)(yrow + k0 + 16 + hi * 8);
        #pragma unroll
        for (int j = 0; j < 8; ++j) {
            Frag b;
            const __bf16* wp = &sWt[ch * 128 + j * 16 + lr][hi * 8];
            b.q[0] = *(const u32x4*)(wp);
            b.q[1] = *(const u32x4*)(wp + 16);
            acc[j] = wmma_bf16(a, b, acc[j]);
        }
    }

    #pragma unroll
    for (int j = 0; j < 8; ++j) {
        const int col = ch * 128 + j * 16 + lr;
        const float bv = Bo[col];
        #pragma unroll
        for (int r = 0; r < 8; ++r) {
            const int row = q0 + rg * 16 + r + hi * 8;
            const size_t idx = ((size_t)bi * N_ + row) * COUT + col;
            OUT[idx] = acc[j][r] + bv + X[idx];
        }
    }
}

// ---------------------------------------------------------------------------
extern "C" void kernel_launch(void* const* d_in, const int* in_sizes, int n_in,
                              void* d_out, int out_size, void* d_ws, size_t ws_size,
                              hipStream_t stream) {
    (void)in_sizes; (void)n_in; (void)out_size; (void)ws_size;
    const float* x    = (const float*)d_in[0];
    const float* w_th = (const float*)d_in[1];
    const float* b_th = (const float*)d_in[2];
    const float* w_ph = (const float*)d_in[3];
    const float* b_ph = (const float*)d_in[4];
    const float* w_g  = (const float*)d_in[5];
    const float* b_g  = (const float*)d_in[6];
    const float* w_o  = (const float*)d_in[7];
    const float* b_o  = (const float*)d_in[8];

    const size_t per = (size_t)B_ * N_ * CI;     // elements per bf16 buffer
    __bf16* th = (__bf16*)d_ws;
    __bf16* ph = th + per;
    __bf16* gT = ph + per;                       // transposed: [B][CI][N]
    __bf16* y  = gT + per;                       // total ws use: ~25.7 MB

    dim3 g1(N_ / QT, B_, 3);
    k_proj<<<g1, 128, 0, stream>>>(x, w_th, w_ph, w_g, b_th, b_ph, b_g, th, ph, gT);
    dim3 g2(N_ / QT, B_);
    const size_t smem = (size_t)(4 * TILE_E + 4 * 16 * KT) * sizeof(__bf16); // 72KB
    k_attn<<<g2, 128, smem, stream>>>(th, ph, gT, y);
    k_out<<<g2, 256, 0, stream>>>(y, w_o, b_o, x, (float*)d_out);
}
